// StableVQVAEModel_46995532153226
// MI455X (gfx1250) — matrix operand order repaired and checked by reading, not compiled, not run on hardware
//
#include <hip/hip_runtime.h>
#include <hip/hip_bf16.h>
#include <math.h>

typedef __attribute__((ext_vector_type(16))) _Float16 v16h;
typedef __attribute__((ext_vector_type(8)))  _Float16 v8h;
typedef __attribute__((ext_vector_type(8)))  float    v8f;

// ---- Generic implicit-GEMM block -------------------------------------------
// 4 waves; 64-row M tile; NT 16-col N subtiles. A fragments are two contiguous
// 8-half runs gathered straight into registers. B is pre-transposed n-major
// [N][Ktot] f16, so each lane's B fragment is one aligned 32B contiguous load.
// No LDS, no barriers.
template <int NT, class FA8, class FS>
__device__ inline void gemm_block(int mBase, int nBase, int Ktot,
                                  const _Float16* __restrict__ Bt,
                                  FA8 loadA8, FS store) {
  const int l = threadIdx.x & 31;
  const int wave = threadIdx.x >> 5;
  const int am   = mBase + wave * 16 + (l & 15);
  const int akhi = (l >> 4) << 4;          // 0 or 16
  const int bn0  = nBase + (l & 15);
  const int bkb  = (l >> 4) << 4;          // 0 or 16
  v8f acc[NT];
#pragma unroll
  for (int t = 0; t < NT; ++t) acc[t] = v8f{};

  const int ksteps = Ktot >> 5;
  for (int ks = 0; ks < ksteps; ++ks) {
    const int k0 = ks * 32;
    const v8h alo = loadA8(am, k0 + akhi);
    const v8h ahi = loadA8(am, k0 + akhi + 8);
    const v16h a = __builtin_shufflevector(alo, ahi,
        0,1,2,3,4,5,6,7,8,9,10,11,12,13,14,15);
#pragma unroll
    for (int t = 0; t < NT; ++t) {
      const v16h b = *(const v16h*)&Bt[(size_t)(bn0 + t * 16) * Ktot + k0 + bkb];
      acc[t] = __builtin_amdgcn_wmma_f32_16x16x32_f16(
          false, a, false, b, (short)0, acc[t], false, false);
    }
  }
  const int mrow = mBase + wave * 16 + ((l >> 4) << 3);
#pragma unroll
  for (int t = 0; t < NT; ++t)
#pragma unroll
    for (int r = 0; r < 8; ++r)
      store(mrow + r, nBase + t * 16 + (l & 15), acc[t][r]);
}

// ---- Weight repack kernels: n-major [N][K] f16, K zero-padded --------------
__global__ void k_repack1(const float* __restrict__ w, _Float16* __restrict__ Bt) {
  const int i = blockIdx.x * 256 + threadIdx.x;          // 64*64
  const int n = i >> 6, k = i & 63;
  float v = 0.0f;
  if (k < 48) { const int c = k >> 4, r = k & 15; v = w[((n * 3 + c) * 4 + (r >> 2)) * 4 + (r & 3)]; }
  Bt[i] = (_Float16)v;
}
__global__ void k_repack2(const float* __restrict__ w, _Float16* __restrict__ Bt) {
  const int i = blockIdx.x * 256 + threadIdx.x;          // 128*1024
  const int n = i >> 10, k = i & 1023, c = k & 63, t = k >> 6;
  Bt[i] = (_Float16)w[((n * 64 + c) * 4 + (t >> 2)) * 4 + (t & 3)];
}
__global__ void k_repack3(const float* __restrict__ w, _Float16* __restrict__ Bt) {
  const int i = blockIdx.x * 256 + threadIdx.x;          // 64*1152
  const int n = i / 1152, k = i % 1152, c = k & 127, t = k >> 7;
  Bt[i] = (_Float16)w[((n * 128 + c) * 3 + (t / 3)) * 3 + (t % 3)];
}
__global__ void k_repackd1(const float* __restrict__ w, _Float16* __restrict__ Bt) {
  const int i = blockIdx.x * 256 + threadIdx.x;          // 128*576
  const int n = i / 576, k = i % 576, c = k & 63, t = k >> 6;
  Bt[i] = (_Float16)w[((n * 64 + c) * 3 + (t / 3)) * 3 + (t % 3)];
}
__global__ void k_repackt1(const float* __restrict__ w, _Float16* __restrict__ Bt) {
  const int i = blockIdx.x * 256 + threadIdx.x;          // 4*64*512
  const int z = i >> 15, rem = i & 32767, n = rem >> 9, k = rem & 511;
  const int k0y = (z >> 1) ? 0 : 1, k0x = (z & 1) ? 0 : 1;
  const int c = k & 127, t = k >> 7;
  const int ky = k0y + 2 * (t >> 1), kx = k0x + 2 * (t & 1);
  Bt[i] = (_Float16)w[((c * 64 + n) * 4 + ky) * 4 + kx];  // (Cin,Cout,4,4)
}
__global__ void k_repackt2(const float* __restrict__ w, _Float16* __restrict__ Bt) {
  const int i = blockIdx.x * 256 + threadIdx.x;          // 4*16*256
  const int z = i >> 12, rem = i & 4095, n = rem >> 8, k = rem & 255;
  const int k0y = (z >> 1) ? 0 : 1, k0x = (z & 1) ? 0 : 1;
  const int c = k & 63, t = k >> 6;
  const int ky = k0y + 2 * (t >> 1), kx = k0x + 2 * (t & 1);
  Bt[i] = (_Float16)((n < 3) ? w[((c * 3 + n) * 4 + ky) * 4 + kx] : 0.0f);  // (64,3,4,4)
}

// ---- Stage kernels ---------------------------------------------------------
// enc conv1: x(16,3,256,256) f32 -> h1 NHWC(16,128,128,64) f16, k4 s2 p1, ReLU
__global__ __launch_bounds__(128) void k_conv1(const float* __restrict__ x,
    const _Float16* __restrict__ Bt, const float* __restrict__ bias,
    _Float16* __restrict__ h1) {
  const int mBase = blockIdx.x * 64;
  auto loadA8 = [&](int m, int k) -> v8h {
    v8h a = v8h{};
    const int ni = m >> 14, rem = m & 16383, oy = rem >> 7, ox = rem & 127;
#pragma unroll
    for (int e = 0; e < 8; ++e) {
      const int kk = k + e;
      if (kk < 48) {
        const int c = kk >> 4, r = kk & 15;
        const int iy = oy * 2 - 1 + (r >> 2), ix = ox * 2 - 1 + (r & 3);
        if ((unsigned)iy < 256u && (unsigned)ix < 256u)
          a[e] = (_Float16)x[((ni * 3 + c) * 256 + iy) * 256 + ix];
      }
    }
    return a;
  };
  auto store = [&](int m, int n, float v) {
    v += bias[n]; v = v > 0.0f ? v : 0.0f;
    const int ni = m >> 14, rem = m & 16383, oy = rem >> 7, ox = rem & 127;
    h1[(((ni * 128 + oy) * 128 + ox) << 6) + n] = (_Float16)v;
  };
  gemm_block<4>(mBase, 0, 64, Bt, loadA8, store);
}

// enc conv2: h1 -> h2 NHWC(16,64,64,128) f16, k4 s2 p1, ReLU (K=1024)
__global__ __launch_bounds__(128) void k_conv2(const _Float16* __restrict__ h1,
    const _Float16* __restrict__ Bt, const float* __restrict__ bias,
    _Float16* __restrict__ h2) {
  const int mBase = blockIdx.x * 64, nBase = blockIdx.y * 64;
  auto loadA8 = [&](int m, int k) -> v8h {
    const int ni = m >> 12, oy = (m >> 6) & 63, ox = m & 63;
    const int c = k & 63, t = k >> 6;
    const int iy = oy * 2 - 1 + (t >> 2), ix = ox * 2 - 1 + (t & 3);
    if ((unsigned)iy >= 128u || (unsigned)ix >= 128u) return v8h{};
    return *(const v8h*)&h1[(((ni * 128 + iy) * 128 + ix) << 6) + c];
  };
  auto store = [&](int m, int n, float v) {
    v += bias[n]; v = v > 0.0f ? v : 0.0f;
    const int ni = m >> 12, oy = (m >> 6) & 63, ox = m & 63;
    h2[(((ni * 64 + oy) * 64 + ox) << 7) + n] = (_Float16)v;
  };
  gemm_block<4>(mBase, nBase, 1024, Bt, loadA8, store);
}

// enc conv3: h2 -> z_e NCHW(16,64,64,64) f32, k3 s1 p1, linear (K=1152)
__global__ __launch_bounds__(128) void k_conv3(const _Float16* __restrict__ h2,
    const _Float16* __restrict__ Bt, const float* __restrict__ bias,
    float* __restrict__ ze) {
  const int mBase = blockIdx.x * 64;
  auto loadA8 = [&](int m, int k) -> v8h {
    const int ni = m >> 12, oy = (m >> 6) & 63, ox = m & 63;
    const int c = k & 127, t = k >> 7;
    const int iy = oy - 1 + (t / 3), ix = ox - 1 + (t % 3);
    if ((unsigned)iy >= 64u || (unsigned)ix >= 64u) return v8h{};
    return *(const v8h*)&h2[(((ni * 64 + iy) * 64 + ix) << 7) + c];
  };
  auto store = [&](int m, int n, float v) {
    v += bias[n];
    const int ni = m >> 12, oy = (m >> 6) & 63, ox = m & 63;
    ze[((ni * 64 + n) * 64 + oy) * 64 + ox] = v;     // NCHW for VQ flatten
  };
  gemm_block<4>(mBase, 0, 1152, Bt, loadA8, store);
}

// VQ: argmin_n (|c_n|^2 - 2 z.c_n) over 512 codes, per 64-elem vector.
// Codebook staged per block in LDS in WMMA B-fragment order (64 KB).
// Writer lanes also scatter the one-hot 1.0f into the (pre-zeroed) encodings.
__global__ __launch_bounds__(128) void k_vq(const float* __restrict__ ze,
    const float* __restrict__ cb, const float* __restrict__ cbn,
    int* __restrict__ idxbuf, unsigned* __restrict__ counts,
    float* __restrict__ enc) {
  __shared__ _Float16 sCB[32 * 2 * 32 * 16];   // [ntile][khalf][lane][e]
  const int wave = threadIdx.x >> 5, l = threadIdx.x & 31;
  const int mBase = blockIdx.x * 64 + wave * 16;
  const int mrow = l & 15, khi = (l >> 4) << 3;

  for (int j = threadIdx.x; j < 2048; j += 128) {
    const int nt = j >> 6, half = (j >> 5) & 1, ll = j & 31;
    const int n  = nt * 16 + (ll & 15);
    const int kb = half * 32 + ((ll >> 4) << 4);
    const float4* src = (const float4*)&cb[n * 64 + kb];
    const float4 f0 = src[0], f1 = src[1], f2 = src[2], f3 = src[3];
    v16h tb;
    tb[0]=(_Float16)f0.x; tb[1]=(_Float16)f0.y; tb[2]=(_Float16)f0.z; tb[3]=(_Float16)f0.w;
    tb[4]=(_Float16)f1.x; tb[5]=(_Float16)f1.y; tb[6]=(_Float16)f1.z; tb[7]=(_Float16)f1.w;
    tb[8]=(_Float16)f2.x; tb[9]=(_Float16)f2.y; tb[10]=(_Float16)f2.z; tb[11]=(_Float16)f2.w;
    tb[12]=(_Float16)f3.x; tb[13]=(_Float16)f3.y; tb[14]=(_Float16)f3.z; tb[15]=(_Float16)f3.w;
    *(v16h*)&sCB[j * 16] = tb;
  }

  v16h a0, a1;
#pragma unroll
  for (int e = 0; e < 16; ++e) {
    const int k = ((l >> 4) << 4) + e;
    a0[e] = (_Float16)ze[(mBase + mrow) * 64 + k];
    a1[e] = (_Float16)ze[(mBase + mrow) * 64 + 32 + k];
  }
  float minv[8]; int mini[8];
#pragma unroll
  for (int r = 0; r < 8; ++r) { minv[r] = 3.4e38f; mini[r] = 0; }
  __syncthreads();

  for (int nt = 0; nt < 32; ++nt) {
    const int n = nt * 16 + (l & 15);
    const v16h b0 = *(const v16h*)&sCB[((nt * 2 + 0) * 32 + l) * 16];
    const v16h b1 = *(const v16h*)&sCB[((nt * 2 + 1) * 32 + l) * 16];
    v8f acc = v8f{};
    acc = __builtin_amdgcn_wmma_f32_16x16x32_f16(false, a0, false, b0, (short)0, acc, false, false);
    acc = __builtin_amdgcn_wmma_f32_16x16x32_f16(false, a1, false, b1, (short)0, acc, false, false);
    const float cn = cbn[n];
#pragma unroll
    for (int r = 0; r < 8; ++r) {
      const float d = cn - 2.0f * acc[r];
      if (d < minv[r] || (d == minv[r] && n < mini[r])) { minv[r] = d; mini[r] = n; }
    }
  }
#pragma unroll
  for (int r = 0; r < 8; ++r) {
#pragma unroll
    for (int off = 8; off > 0; off >>= 1) {
      const float ov = __shfl_xor(minv[r], off, 16);
      const int   oi = __shfl_xor(mini[r], off, 16);
      if (ov < minv[r] || (ov == minv[r] && oi < mini[r])) { minv[r] = ov; mini[r] = oi; }
    }
    if ((l & 15) == 0) {
      const int row = mBase + khi + r;
      idxbuf[row] = mini[r];
      enc[(long long)row * 512 + mini[r]] = 1.0f;
      atomicAdd(&counts[mini[r]], 1u);
    }
  }
}

// quantized gather (NHWC f16 for decoder) + sum((q-z)^2)
__global__ __launch_bounds__(256) void k_gatherq(const float* __restrict__ ze,
    const float* __restrict__ cb, const int* __restrict__ idxbuf,
    _Float16* __restrict__ q, float* __restrict__ loss_acc) {
  const int i = blockIdx.x * 256 + threadIdx.x;         // 0 .. 16*64*64*64-1
  const int v = i >> 6, wcol = i & 63;
  const float qv = cb[idxbuf[v] * 64 + wcol];
  const float d = qv - ze[i];
  float s = d * d;
  const int ni = v >> 12, c = (v >> 6) & 63, h = v & 63;
  q[(((ni * 64 + h) * 64 + wcol) << 6) + c] = (_Float16)qv;
#pragma unroll
  for (int off = 16; off > 0; off >>= 1) s += __shfl_down(s, off, 32);
  if ((threadIdx.x & 31) == 0) atomicAdd(loss_acc, s);
}

// dec conv: q NHWC(16,64,64,64) -> d1 NHWC(16,64,64,128), k3 p1, ReLU (K=576)
__global__ __launch_bounds__(128) void k_dconv1(const _Float16* __restrict__ q,
    const _Float16* __restrict__ Bt, const float* __restrict__ bias,
    _Float16* __restrict__ d1) {
  const int mBase = blockIdx.x * 64, nBase = blockIdx.y * 64;
  auto loadA8 = [&](int m, int k) -> v8h {
    const int ni = m >> 12, oy = (m >> 6) & 63, ox = m & 63;
    const int c = k & 63, t = k >> 6;
    const int iy = oy - 1 + (t / 3), ix = ox - 1 + (t % 3);
    if ((unsigned)iy >= 64u || (unsigned)ix >= 64u) return v8h{};
    return *(const v8h*)&q[(((ni * 64 + iy) * 64 + ix) << 6) + c];
  };
  auto store = [&](int m, int n, float v) {
    v += bias[n]; v = v > 0.0f ? v : 0.0f;
    const int ni = m >> 12, oy = (m >> 6) & 63, ox = m & 63;
    d1[(((ni * 64 + oy) * 64 + ox) << 7) + n] = (_Float16)v;
  };
  gemm_block<4>(mBase, nBase, 576, Bt, loadA8, store);
}

// convT1: d1 -> d2 NHWC(16,128,128,64), k4 s2 p1, ReLU. gridDim.z = parity class
__global__ __launch_bounds__(128) void k_tconv1(const _Float16* __restrict__ d1,
    const _Float16* __restrict__ Bt, const float* __restrict__ bias,
    _Float16* __restrict__ d2) {
  const int mBase = blockIdx.x * 64;
  const int py = (blockIdx.z >> 1) & 1, px = blockIdx.z & 1;
  const int k0y = py ? 0 : 1, k0x = px ? 0 : 1;   // o=2i-1+k -> k parity = (o+1)&1
  const _Float16* Bz = Bt + (size_t)blockIdx.z * (64 * 512);
  auto loadA8 = [&](int m, int k) -> v8h {
    const int ni = m >> 12, jy = (m >> 6) & 63, jx = m & 63;
    const int oy = jy * 2 + py, ox = jx * 2 + px;
    const int c = k & 127, t = k >> 7;
    const int iy = (oy + 1 - (k0y + 2 * (t >> 1))) >> 1;
    const int ix = (ox + 1 - (k0x + 2 * (t & 1))) >> 1;
    if ((unsigned)iy >= 64u || (unsigned)ix >= 64u) return v8h{};
    return *(const v8h*)&d1[(((ni * 64 + iy) * 64 + ix) << 7) + c];
  };
  auto store = [&](int m, int n, float v) {
    v += bias[n]; v = v > 0.0f ? v : 0.0f;
    const int ni = m >> 12, jy = (m >> 6) & 63, jx = m & 63;
    d2[(((ni * 128 + jy * 2 + py) * 128 + jx * 2 + px) << 6) + n] = (_Float16)v;
  };
  gemm_block<4>(mBase, 0, 512, Bz, loadA8, store);
}

// convT2: d2 -> x_recon NCHW(16,3,256,256) f32, k4 s2 p1, tanh (K=256, N=3)
__global__ __launch_bounds__(128) void k_tconv2(const _Float16* __restrict__ d2,
    const _Float16* __restrict__ Bt, const float* __restrict__ bias,
    float* __restrict__ xr) {
  const int mBase = blockIdx.x * 64;
  const int py = (blockIdx.z >> 1) & 1, px = blockIdx.z & 1;
  const int k0y = py ? 0 : 1, k0x = px ? 0 : 1;
  const _Float16* Bz = Bt + (size_t)blockIdx.z * (16 * 256);
  auto loadA8 = [&](int m, int k) -> v8h {
    const int ni = m >> 14, jy = (m >> 7) & 127, jx = m & 127;
    const int oy = jy * 2 + py, ox = jx * 2 + px;
    const int c = k & 63, t = k >> 6;
    const int iy = (oy + 1 - (k0y + 2 * (t >> 1))) >> 1;
    const int ix = (ox + 1 - (k0x + 2 * (t & 1))) >> 1;
    if ((unsigned)iy >= 128u || (unsigned)ix >= 128u) return v8h{};
    return *(const v8h*)&d2[(((ni * 128 + iy) * 128 + ix) << 6) + c];
  };
  auto store = [&](int m, int n, float v) {
    if (n >= 3) return;
    const int ni = m >> 14, jy = (m >> 7) & 127, jx = m & 127;
    xr[((ni * 3 + n) * 256 + jy * 2 + py) * 256 + jx * 2 + px] = tanhf(v + bias[n]);
  };
  gemm_block<1>(mBase, 0, 256, Bz, loadA8, store);
}

// ---- small helpers ---------------------------------------------------------
// cbnorm + counters + zero the unaligned head/tail scalars of encodings
__global__ void k_init(const float* __restrict__ cb, float* __restrict__ cbn,
                       unsigned* __restrict__ counts, float* __restrict__ loss_acc,
                       float* __restrict__ enc) {
  const int i = blockIdx.x * 256 + threadIdx.x;
  if (i < 512) {
    float s = 0.0f;
    for (int d = 0; d < 64; ++d) { const float v = cb[i * 64 + d]; s += v * v; }
    cbn[i] = s; counts[i] = 0u;
  }
  if (i == 0) *loss_acc = 0.0f;
  if (i < 2) { enc[i] = 0.0f; enc[33554430LL + i] = 0.0f; }
}

__global__ void k_zero4(float4* __restrict__ p, long long n4) {
  long long i = (long long)blockIdx.x * 256 + threadIdx.x;
  const long long st = (long long)gridDim.x * 256;
  const float4 z = make_float4(0.0f, 0.0f, 0.0f, 0.0f);
  for (; i < n4; i += st) p[i] = z;
}

__global__ void k_fin(const float* __restrict__ loss_acc,
                      const unsigned* __restrict__ counts,
                      float* __restrict__ out_loss, float* __restrict__ out_perp) {
  const float m = *loss_acc / 4194304.0f;        // 16*64*64*64 elements
  *out_loss = 1.25f * m;                         // 0.25*e_latent + q_latent (equal fwd)
  float H = 0.0f;
  for (int k = 0; k < 512; ++k) {
    const float p = (float)counts[k] / 65536.0f;
    H += p * logf(p + 1e-10f);
  }
  *out_perp = expf(-H);
}

// ---- launch ----------------------------------------------------------------
extern "C" void kernel_launch(void* const* d_in, const int* in_sizes, int n_in,
                              void* d_out, int out_size, void* d_ws, size_t ws_size,
                              hipStream_t stream) {
  const float* x    = (const float*)d_in[0];
  const float* ew1  = (const float*)d_in[1];  const float* eb1 = (const float*)d_in[2];
  const float* ew2  = (const float*)d_in[3];  const float* eb2 = (const float*)d_in[4];
  const float* ew3  = (const float*)d_in[5];  const float* eb3 = (const float*)d_in[6];
  const float* cb   = (const float*)d_in[7];
  const float* dw1  = (const float*)d_in[8];  const float* db1  = (const float*)d_in[9];
  const float* dwt1 = (const float*)d_in[10]; const float* dbt1 = (const float*)d_in[11];
  const float* dwt2 = (const float*)d_in[12]; const float* dbt2 = (const float*)d_in[13];

  char* ws = (char*)d_ws;
  _Float16* h1  = (_Float16*)(ws + 0);            // 33.5 MB (aliases d2)
  _Float16* d2  = h1;
  _Float16* h2  = (_Float16*)(ws + 33554432);     // 16.8 MB (aliases d1)
  _Float16* d1  = h2;
  float*    ze  = (float*)(ws + 50331648);        // 16.8 MB
  _Float16* q   = (_Float16*)(ws + 67108864);     // 8.4 MB
  int*      idx = (int*)(ws + 75497472);
  unsigned* cnt = (unsigned*)(ws + 75759616);
  float*    cbn = (float*)(ws + 75761664);
  float*    lac = (float*)(ws + 75763712);
  _Float16* B1t = (_Float16*)(ws + 75763968);     // 64*64
  _Float16* B2t = (_Float16*)(ws + 75772160);     // 128*1024
  _Float16* B3t = (_Float16*)(ws + 76034304);     // 64*1152
  _Float16* Bd1 = (_Float16*)(ws + 76181760);     // 128*576
  _Float16* Bt1 = (_Float16*)(ws + 76329216);     // 4*64*512
  _Float16* Bt2 = (_Float16*)(ws + 76591360);     // 4*16*256

  float* out  = (float*)d_out;
  float* xr   = out;                  // 16*3*256*256
  float* losp = out + 3145728;
  float* perp = out + 3145729;
  float* enc  = out + 3145730;        // 65536*512 (head/tail handled in k_init)

  k_repack1 <<<16, 256, 0, stream>>>(ew1, B1t);
  k_repack2 <<<512, 256, 0, stream>>>(ew2, B2t);
  k_repack3 <<<288, 256, 0, stream>>>(ew3, B3t);
  k_repackd1<<<288, 256, 0, stream>>>(dw1, Bd1);
  k_repackt1<<<512, 256, 0, stream>>>(dwt1, Bt1);
  k_repackt2<<<64, 256, 0, stream>>>(dwt2, Bt2);
  k_init    <<<2, 256, 0, stream>>>(cb, cbn, cnt, lac, enc);
  // aligned float4 body: [enc+2, enc+2 + 4*8388607)
  k_zero4   <<<4096, 256, 0, stream>>>((float4*)(enc + 2), 8388607LL);

  k_conv1  <<<dim3(4096), 128, 0, stream>>>(x, B1t, eb1, h1);
  k_conv2  <<<dim3(1024, 2), 128, 0, stream>>>(h1, B2t, eb2, h2);
  k_conv3  <<<dim3(1024), 128, 0, stream>>>(h2, B3t, eb3, ze);
  k_vq     <<<1024, 128, 0, stream>>>(ze, cb, cbn, idx, cnt, enc);
  k_gatherq<<<16384, 256, 0, stream>>>(ze, cb, idx, q, lac);
  k_dconv1 <<<dim3(1024, 2), 128, 0, stream>>>(q, Bd1, db1, d1);
  k_tconv1 <<<dim3(1024, 1, 4), 128, 0, stream>>>(d1, Bt1, dbt1, d2);
  k_tconv2 <<<dim3(4096, 1, 4), 128, 0, stream>>>(d2, Bt2, dbt2, xr);
  k_fin    <<<1, 1, 0, stream>>>(lac, cnt, losp, perp);
}
